// IGM_86577950753226
// MI455X (gfx1250) — matrix operand order repaired and checked by reading, not compiled, not run on hardware
//
#include <hip/hip_runtime.h>
#include <hip/hip_bf16.h>
#include <math.h>

// ---------------------------------------------------------------------------
// Problem constants (from reference)
// ---------------------------------------------------------------------------
constexpr int G    = 512;
constexpr int NPG  = 128;
constexpr int EPG  = 512;
constexpr int EMB  = 128;
constexpr int E    = G * EPG;      // 262144 edges
constexpr int NRES = EPG / 4;      // 128 causal edges per group

// d_out layout (floats), in reference return order
constexpr int O_W   = 0;                        // w            [E]
constexpr int O_CW  = O_W   + E;                // causal_w     [G*128]
constexpr int O_SW  = O_CW  + G * NRES;         // spu_w        [G*384]
constexpr int O_CA  = O_SW  + G * (EPG - NRES); // causal_attr
constexpr int O_SA  = O_CA  + G * NRES;
constexpr int O_CEI = O_SA  + G * (EPG - NRES); // causal_ei [2][G*128]
constexpr int O_SEI = O_CEI + 2 * G * NRES;     // spu_ei    [2][G*384]

// ---------------------------------------------------------------------------
// WMMA / TDM vector types (CDNA5 wave32)
// ---------------------------------------------------------------------------
typedef __attribute__((ext_vector_type(16))) __bf16        bf16x16;
typedef __attribute__((ext_vector_type(8)))  float         f32x8;
typedef __attribute__((ext_vector_type(4)))  unsigned int  uint32x4;
typedef __attribute__((ext_vector_type(8)))  int           int32x8;
typedef __attribute__((ext_vector_type(4)))  int           int32x4;

struct U8x4 { uint4 a, b; };   // 32 bytes -> bit_cast to bf16x16 (B fragment)

// LDS tile: 256 K-rows x 512 N-cols of bf16, rows padded to 260 dwords
// (+4 dwords skew per row -> 2-way worst-case bank conflict on b128 reads;
//  260*4 bytes keeps every row 16B-aligned so ds_load_b128 stays aligned)
constexpr int LDS_ROW_DW = 260;
constexpr int LDS_BYTES  = 256 * LDS_ROW_DW * 4;   // 266,240 B <= 320 KB/WGP

// ---------------------------------------------------------------------------
// Deterministic hash -> uniform(0,1) -> Gumbel (stand-in for jax threefry)
// ---------------------------------------------------------------------------
__device__ __forceinline__ float hash_uniform(unsigned x) {
  x ^= x >> 16; x *= 0x7feb352du;
  x ^= x >> 15; x *= 0x846ca68bu;
  x ^= x >> 16;
  return ((float)x + 0.5f) * (1.0f / 4294967296.0f);
}
__device__ __forceinline__ float gumbel(unsigned x) {
  return -logf(-logf(hash_uniform(x)));
}

// ---------------------------------------------------------------------------
// Kernel 0: W1 f32 -> bf16 (256x512, row-major) into workspace
// ---------------------------------------------------------------------------
__global__ void cvt_w1_bf16(const float* __restrict__ w1, __bf16* __restrict__ w1b) {
  int i = blockIdx.x * 256 + threadIdx.x;   // launched with exactly 2*EMB*4*EMB elems
  w1b[i] = (__bf16)w1[i];
}

// ---------------------------------------------------------------------------
// Kernel 1: fused edge-MLP via v_wmma_f32_16x16x32_bf16 + TDM-staged W1
//   block = 256 threads (8 waves), each block handles 128 edges.
//   Each wave: 16 edges (M), loops 32 N-tiles x 8 K-tiles of WMMA, with the
//   B fragments software-pipelined one step ahead of the WMMA consuming them.
// ---------------------------------------------------------------------------
__global__ void edge_mlp_wmma(const float* __restrict__ h,
                              const int*   __restrict__ ei,
                              const __bf16* __restrict__ w1b,
                              const float* __restrict__ b1,
                              const float* __restrict__ w2,
                              const float* __restrict__ b2,
                              float* __restrict__ w_out) {
  extern __shared__ __align__(16) unsigned int lds[];  // 256 x 260 dwords
  const int tid = threadIdx.x;

  // --- TDM: DMA full bf16 W1 (256 rows x 512 cols, 256 KB) into LDS.
  // D# pad fields recreate the 260-dword padded rows in hardware:
  //   pad_interval=7 -> pad after every 256 dwords (= one 512-elem bf16 row)
  //   pad_amount  =3 -> insert 4 dwords of padding
  if (tid < 32) {                       // wave 0 issues the descriptor
    const unsigned ldsBase = __builtin_amdgcn_groupstaticsize(); // dyn-LDS base
    const unsigned long long ga = (unsigned long long)(uintptr_t)w1b;

    uint32x4 g0;
    g0[0] = 1u;                                   // count=1, user descriptor
    g0[1] = ldsBase;                              // lds_addr (bytes)
    g0[2] = (unsigned)(ga & 0xFFFFFFFFu);         // global_addr[31:0]
    g0[3] = (unsigned)((ga >> 32) & 0x01FFFFFFu)  // global_addr[56:32]
          | (2u << 30);                           // type=2 ("image")

    int32x8 g1;
    g1[0] = (1 << 16)      // data_size = 2 bytes (bf16)
          | (1 << 20)      // pad_enable
          | (7 << 22)      // pad_interval: 256 dwords
          | (3 << 25);     // pad_amount: 4 dwords
    g1[1] = (int)(512u << 16);                    // tensor_dim0 = 512 (lo 16)
    g1[2] = (int)((512u >> 16) | (256u << 16));   // dim0 hi | tensor_dim1 lo
    g1[3] = (int)((256u >> 16) | (512u << 16));   // dim1 hi | tile_dim0 = 512
    g1[4] = 256;                                  // tile_dim1=256, tile_dim2=0
    g1[5] = 512;                                  // tensor_dim0_stride lo32
    g1[6] = 0;                                    // stride hi | dim1_stride lo
    g1[7] = 0;

    int32x4 gz4 = {0, 0, 0, 0};                   // 2D tensor: groups 2/3 unused
    int32x8 gz8 = {0, 0, 0, 0, 0, 0, 0, 0};
    __builtin_amdgcn_tensor_load_to_lds(g0, g1, gz4, gz4, gz8, 0);
    __builtin_amdgcn_s_wait_tensorcnt(0);
  }
  __syncthreads();

  const int lane  = tid & 31;
  const int wave  = tid >> 5;
  const int m     = lane & 15;     // M row within 16x16 tile
  const int hi    = lane >> 4;     // lane half (selects K sub-blocks)
  const int eBase = blockIdx.x * 128 + wave * 16;
  const int e     = eBase + m;
  const int row   = ei[e];
  const int col   = ei[E + e];

  // --- A fragments: 8 K-tiles of the 16x256 bf16 edge_rep tile.
  // ISA A layout (16-bit 16x32): lanes 0-15 hold K 0-7 & 16-23,
  // lanes 16-31 hold K 8-15 & 24-31 (rel. to tile base). 128%32==0 so each
  // K-tile lies wholly in h[row] (kk<4) or h[col] (kk>=4).
  bf16x16 afrag[8];
#pragma unroll
  for (int kk = 0; kk < 8; ++kk) {
    const float* src = (kk < 4) ? (h + (size_t)row * EMB + kk * 32)
                                : (h + (size_t)col * EMB + (kk - 4) * 32);
    const float* s0 = src + hi * 8;        // K = base + hi*8 + 0..7
    const float* s1 = src + hi * 8 + 16;   // K = base + hi*8 + 16..23
    float4 q0 = *(const float4*)(s0);
    float4 q1 = *(const float4*)(s0 + 4);
    float4 q2 = *(const float4*)(s1);
    float4 q3 = *(const float4*)(s1 + 4);
    bf16x16 a;
    a[0]=(__bf16)q0.x; a[1]=(__bf16)q0.y; a[2]=(__bf16)q0.z;  a[3]=(__bf16)q0.w;
    a[4]=(__bf16)q1.x; a[5]=(__bf16)q1.y; a[6]=(__bf16)q1.z;  a[7]=(__bf16)q1.w;
    a[8]=(__bf16)q2.x; a[9]=(__bf16)q2.y; a[10]=(__bf16)q2.z; a[11]=(__bf16)q2.w;
    a[12]=(__bf16)q3.x;a[13]=(__bf16)q3.y;a[14]=(__bf16)q3.z; a[15]=(__bf16)q3.w;
    afrag[kk] = a;
  }

  // --- GEMM + relu + second-layer dot, fused over 32 N-tiles ---
  const uint4* lds4 = (const uint4*)lds;   // 16B-aligned b128 view of LDS

  // B layout (bf16 32x16): lane%16 = N column; elements follow the A K order
  // (K = hi*8 + 0..7, then 16 + hi*8 + 0..7).
  auto loadB = [&](int n, int kk) -> bf16x16 {
    const __bf16* lb = (const __bf16*)lds;
    bf16x16 b;
#pragma unroll
    for (int s = 0; s < 16; ++s) {
      const int k = (s < 8) ? (hi * 8 + s) : (16 + hi * 8 + (s - 8));
      b[s] = lb[(kk * 32 + k) * (LDS_ROW_DW * 2) + n * 16 + m];
    }
    return b;
  };

  float acc[8];
#pragma unroll
  for (int r = 0; r < 8; ++r) acc[r] = 0.0f;

  bf16x16 bcur = loadB(0, 0);              // pipeline prologue
  for (int n = 0; n < 32; ++n) {
    const int ncol = n * 16 + m;           // this lane's N column (C layout)
    const float bias = b1[ncol];
    const float w2v  = w2[ncol];
    f32x8 c;
#pragma unroll
    for (int r = 0; r < 8; ++r) c[r] = bias;   // fold +b1 into accumulator

#pragma unroll
    for (int kk = 0; kk < 8; ++kk) {
      // Prefetch next B fragment before the WMMA consuming the current one;
      // wrap the final index so the epilogue load is branch-free (row 0 reread).
      bf16x16 bnext = (kk < 7) ? loadB(n, kk + 1) : loadB((n + 1) & 31, 0);
      c = __builtin_amdgcn_wmma_f32_16x16x32_bf16(
              false, afrag[kk], false, bcur, (short)0, c, false, false);
      bcur = bnext;
    }

#pragma unroll
    for (int r = 0; r < 8; ++r) acc[r] += fmaxf(c[r], 0.0f) * w2v;
  }

  // --- reduce columns across each 16-lane half: lane0 -> edge r, lane16 -> edge 8+r
  const float bias2 = b2[0];
#pragma unroll
  for (int r = 0; r < 8; ++r) {
    float v = acc[r];
#pragma unroll
    for (int off = 1; off < 16; off <<= 1) v += __shfl_xor(v, off, 16);
    if ((lane & 15) == 0) {
      const int eo = eBase + hi * 8 + r;
      const float s  = 1.0f / (1.0f + expf(-(v + bias2)));
      const float g1 = gumbel((unsigned)eo * 2u + 0x9E3779B9u);
      const float g2 = gumbel((unsigned)eo * 2u + 0x85EBCA6Bu);
      const float p1 = logf(s + 1e-9f) + g1;
      const float p2 = logf(1.0f - s + 1e-9f) + g2;
      w_out[eo] = 1.0f / (1.0f + expf(-(p1 - p2)));
    }
  }
}

// ---------------------------------------------------------------------------
// Kernel 2: per-group bitonic sort (descending, 512 keys) + fused selection
//   gather of causal/spurious w, attr, edge_index slices.
// ---------------------------------------------------------------------------
__global__ void sort_select(const float* __restrict__ edge_attr,
                            const int*   __restrict__ ei,
                            float* out) {
  __shared__ float key[EPG];
  __shared__ int   idx[EPG];
  const int g = blockIdx.x, tid = threadIdx.x;
  const float* w = out + O_W;

  for (int i = tid; i < EPG; i += 256) { key[i] = w[g * EPG + i]; idx[i] = i; }
  __syncthreads();

  for (int k = 2; k <= EPG; k <<= 1) {
    for (int j = k >> 1; j > 0; j >>= 1) {
      const int i = 2 * tid - (tid & (j - 1));   // bit j clear
      const int p = i + j;
      const bool desc = ((i & k) == 0);
      const float ki = key[i], kp = key[p];
      const bool sw = desc ? (ki < kp) : (ki > kp);
      if (sw) {
        key[i] = kp; key[p] = ki;
        int t = idx[i]; idx[i] = idx[p]; idx[p] = t;
      }
      __syncthreads();
    }
  }

  const int nCau = G * NRES, nSpu = G * (EPG - NRES);
  for (int i = tid; i < EPG; i += 256) {
    const int glob = g * EPG + idx[i];
    const float wv = key[i];
    const float av = edge_attr[glob];
    const float e0 = (float)ei[glob];
    const float e1 = (float)ei[E + glob];
    if (i < NRES) {
      const int s = g * NRES + i;
      out[O_CW + s] = wv;
      out[O_CA + s] = av;
      out[O_CEI + s]        = e0;
      out[O_CEI + nCau + s] = e1;
    } else {
      const int s = g * (EPG - NRES) + (i - NRES);
      out[O_SW + s] = wv;
      out[O_SA + s] = av;
      out[O_SEI + s]        = e0;
      out[O_SEI + nSpu + s] = e1;
    }
  }
}

// ---------------------------------------------------------------------------
extern "C" void kernel_launch(void* const* d_in, const int* in_sizes, int n_in,
                              void* d_out, int out_size, void* d_ws, size_t ws_size,
                              hipStream_t stream) {
  const float* h    = (const float*)d_in[0];
  const int*   ei   = (const int*)  d_in[1];
  const float* attr = (const float*)d_in[2];
  const float* W1   = (const float*)d_in[3];
  const float* b1   = (const float*)d_in[4];
  const float* W2   = (const float*)d_in[5];
  const float* b2   = (const float*)d_in[6];
  float*   out  = (float*)d_out;
  __bf16*  w1b  = (__bf16*)d_ws;          // 256 KB of workspace

  cvt_w1_bf16<<<(2 * EMB * 4 * EMB) / 256, 256, 0, stream>>>(W1, w1b);
  edge_mlp_wmma<<<E / 128, 256, LDS_BYTES, stream>>>(h, ei, w1b, b1, W2, b2,
                                                     out + O_W);
  sort_select<<<G, 256, 0, stream>>>(attr, ei, out);
}
